// WorkingMemory_66185446031743
// MI455X (gfx1250) — matrix operand order repaired and checked by reading, not compiled, not run on hardware
//
#include <hip/hip_runtime.h>
#include <hip/hip_bf16.h>

typedef __attribute__((ext_vector_type(16))) _Float16 v16h;
typedef __attribute__((ext_vector_type(8)))  float    v8f;
typedef __attribute__((ext_vector_type(4)))  float    v4f;

#define D 128
#define NSLOT 8
#define OUTD 1288                 // 2D + N*D + N
#define ROWS_PER_BLOCK 64
#define ROWS_PER_WAVE 16
#define SROW 132                  // padded LDS row stride (floats), 16B aligned

__device__ __forceinline__ float wred32(float v) {
#pragma unroll
    for (int m = 16; m > 0; m >>= 1) v += __shfl_xor(v, m, 32);
    return v;
}

__device__ __forceinline__ float dot4(v4f a, v4f b) {
    return a.x * b.x + a.y * b.y + a.z * b.z + a.w * b.w;
}

// ---------------------------------------------------------------------------
// Prep: M = Wq^T @ Wk (f16), WfT = Wf^T (f16).  128x128 each -> 64 KB of ws.
// ---------------------------------------------------------------------------
__global__ void wm_prep(const float* __restrict__ Wq, const float* __restrict__ Wk,
                        const float* __restrict__ Wf,
                        _Float16* __restrict__ wsM, _Float16* __restrict__ wsWfT)
{
    int t = blockIdx.x * blockDim.x + threadIdx.x;   // 0 .. 16383
    int k = t >> 7, n = t & 127;
    float acc = 0.0f;
#pragma unroll 8
    for (int e = 0; e < D; ++e) acc += Wq[e * D + k] * Wk[e * D + n];
    wsM[t]   = (_Float16)acc;                // M[k][n]   = sum_e Wq[e,k]*Wk[e,n]
    wsWfT[t] = (_Float16)Wf[n * D + k];      // WfT[k][n] = Wf[n,k]
}

// ---------------------------------------------------------------------------
// Main: 4 waves / block, 16 batch rows per wave.
// ---------------------------------------------------------------------------
__global__ __launch_bounds__(128)
void wm_main(const float* __restrict__ state, const float* __restrict__ slots,
             const float* __restrict__ acts,  const float* __restrict__ Wg_w,
             const float* __restrict__ Wg_b,
             const _Float16* __restrict__ wsM, const _Float16* __restrict__ wsWfT,
             float* __restrict__ out)
{
    __shared__ __align__(32) _Float16 sW[D * D];               // 32 KB weight tile
    __shared__ __align__(16) float sState[ROWS_PER_BLOCK * SROW];
    __shared__ __align__(16) float sQK[ROWS_PER_BLOCK * SROW]; // qk, later reused for s
    __shared__ float sWg[2 * D];
    __shared__ float sActs[ROWS_PER_BLOCK * NSLOT];
    __shared__ float sWgb;

    const int tid  = threadIdx.x;
    const int wave = tid >> 5;
    const int lane = tid & 31;
    const int blockRow0 = blockIdx.x * ROWS_PER_BLOCK;
    const int wr0 = wave * ROWS_PER_WAVE;
    const int gr0 = blockRow0 + wr0;

    // ---- stage fused weight M (f16): default RT policy -> stays L2-hot ----
    {
        const uint4* src = (const uint4*)wsM;
        uint4* dst = (uint4*)sW;
#pragma unroll
        for (int i = 0; i < 16; ++i) dst[tid + 128 * i] = src[tid + 128 * i];
    }
    // ---- stage state rows (coalesced, single-use stream -> NT) ----
    for (int i = tid; i < ROWS_PER_BLOCK * (D / 4); i += 128) {
        int r = i >> 5, c = i & 31;
        v4f v = __builtin_nontemporal_load(
            (const v4f*)(state + (size_t)(blockRow0 + r) * D) + c);
        *(v4f*)&sState[r * SROW + c * 4] = v;
    }
    // ---- stage activations ----
    for (int i = tid; i < ROWS_PER_BLOCK * NSLOT / 4; i += 128) {
        v4f v = __builtin_nontemporal_load((const v4f*)(acts + (size_t)blockRow0 * NSLOT) + i);
        ((v4f*)sActs)[i] = v;
    }
    sWg[tid] = Wg_w[tid];
    sWg[tid + 128] = Wg_w[tid + 128];
    if (tid == 0) sWgb = Wg_b[0];
    __syncthreads();

    // ================= GEMM1: qk = state @ M (f16 WMMA, f32 accum) ==========
    {
        v8f zero = {};
        v8f acc[8];
#pragma unroll
        for (int nt = 0; nt < 8; ++nt) acc[nt] = zero;
        const int arow = wr0 + (lane & 15);
#pragma unroll
        for (int kk = 0; kk < 4; ++kk) {
            const int kbase = kk * 32 + ((lane >= 16) ? 8 : 0);
            const float* ap = &sState[arow * SROW + kbase];
            v16h a;
#pragma unroll
            for (int e = 0; e < 8; ++e) a[e] = (_Float16)ap[e];
#pragma unroll
            for (int e = 0; e < 8; ++e) a[8 + e] = (_Float16)ap[16 + e];
            const _Float16* bp = &sW[(kk * 32 + ((lane >= 16) ? 8 : 0)) * D + (lane & 15)];
#pragma unroll
            for (int nt = 0; nt < 8; ++nt) {
                v16h b;
#pragma unroll
                for (int e = 0; e < 8; ++e) b[e] = bp[e * D + nt * 16];
#pragma unroll
                for (int e = 0; e < 8; ++e) b[8 + e] = bp[(16 + e) * D + nt * 16];
                acc[nt] = __builtin_amdgcn_wmma_f32_16x16x32_f16(
                    false, a, false, b, (short)0, acc[nt], false, false);
            }
        }
        const int cn = lane & 15;
        const int mb = (lane >> 4) * 8;
#pragma unroll
        for (int nt = 0; nt < 8; ++nt)
#pragma unroll
            for (int j = 0; j < 8; ++j)
                sQK[(wr0 + mb + j) * SROW + nt * 16 + cn] = acc[nt][j];
    }
    __syncthreads();

    // ================= per-row: one pass over the 4 KB slot tile ============
    for (int r = 0; r < ROWS_PER_WAVE; ++r) {
        const int wrow = wr0 + r;
        const size_t gb = (size_t)(gr0 + r);
        const float* slotp = slots + gb * (NSLOT * D) + lane * 4;

        // prefetch next row's slot tile while this row computes
        if (r + 1 < ROWS_PER_WAVE) {
            const float* np = slotp + NSLOT * D;
#pragma unroll
            for (int n = 0; n < NSLOT; ++n) __builtin_prefetch(np + n * D, 0, 3);
        }

        v4f sl[NSLOT];
#pragma unroll
        for (int n = 0; n < NSLOT; ++n)
            sl[n] = __builtin_nontemporal_load((const v4f*)(slotp + n * D));
        const v4f st = *(const v4f*)&sState[wrow * SROW + lane * 4];
        const v4f qk = *(const v4f*)&sQK[wrow * SROW + lane * 4];
        float a0[NSLOT];
#pragma unroll
        for (int n = 0; n < NSLOT; ++n) a0[n] = sActs[wrow * NSLOT + n];

        // 25 wave reductions: |state|^2, and per-slot attn dot / |slot|^2 / sim dot
        float attn[NSLOT], nrm2[NSLOT], simd[NSLOT];
        float stn2 = wred32(dot4(st, st));
#pragma unroll
        for (int n = 0; n < NSLOT; ++n) {
            attn[n] = wred32(dot4(qk, sl[n]));
            nrm2[n] = wred32(dot4(sl[n], sl[n]));
            simd[n] = wred32(dot4(st, sl[n]));
        }

        // ---- read: softmax over activated attention, weighted slot sum ----
        float logits[NSLOT], mx = -1e30f;
#pragma unroll
        for (int n = 0; n < NSLOT; ++n) {
            logits[n] = attn[n] * 0.08838834764831845f * a0[n];   // 1/sqrt(128)
            mx = fmaxf(mx, logits[n]);
        }
        float se = 0.f;
#pragma unroll
        for (int n = 0; n < NSLOT; ++n) { logits[n] = __expf(logits[n] - mx); se += logits[n]; }
        float inv = 1.f / se;
        v4f ro = {0.f, 0.f, 0.f, 0.f};
#pragma unroll
        for (int n = 0; n < NSLOT; ++n) ro += sl[n] * (logits[n] * inv);
        __builtin_nontemporal_store(ro, (v4f*)(out + gb * OUTD + lane * 4));

        // ---- write decision: cosine novelty + gate ----
        float istn = 1.f / fmaxf(sqrtf(stn2), 1e-12f);
        float na[NSLOT], maxsim = -1e30f;
        v4f wm = {0.f, 0.f, 0.f, 0.f};
#pragma unroll
        for (int n = 0; n < NSLOT; ++n) {
            na[n] = a0[n] * 0.95f;
            float isl = 1.f / fmaxf(sqrtf(nrm2[n]), 1e-12f);
            float sim = simd[n] * istn * isl;
            sim = (na[n] > 0.01f) ? sim : 0.f;
            maxsim = fmaxf(maxsim, sim);
            wm += sl[n] * na[n];
        }
        wm *= 0.125f;
        float novelty = 1.f - fminf(fmaxf(maxsim, 0.f), 1.f);
        const v4f g0 = *(const v4f*)&sWg[lane * 4];
        const v4f g1 = *(const v4f*)&sWg[128 + lane * 4];
        float gate = 1.f / (1.f + __expf(-(wred32(dot4(st, g0) + dot4(wm, g1)) + sWgb)));
        bool should = (novelty > 0.3f) || (gate > 0.5f);
        int weakest = 0; float bmin = na[0];
#pragma unroll
        for (int n = 1; n < NSLOT; ++n) if (na[n] < bmin) { bmin = na[n]; weakest = n; }

        // ---- scatter write + new acts + summary softmax input s ----
        float nao[NSLOT], mx2 = -1e30f;
#pragma unroll
        for (int n = 0; n < NSLOT; ++n) {
            nao[n] = (should && (n == weakest)) ? 1.f : na[n];
            mx2 = fmaxf(mx2, nao[n]);
        }
        float se2 = 0.f, sw[NSLOT];
#pragma unroll
        for (int n = 0; n < NSLOT; ++n) { sw[n] = __expf(nao[n] - mx2); se2 += sw[n]; }
        float inv2 = 1.f / se2;
        v4f ssum = {0.f, 0.f, 0.f, 0.f};
#pragma unroll
        for (int n = 0; n < NSLOT; ++n) {
            bool wsel = should && (n == weakest);
            v4f v = wsel ? st : sl[n];
            __builtin_nontemporal_store(v, (v4f*)(out + gb * OUTD + 2 * D + n * D + lane * 4));
            ssum += v * (sw[n] * inv2);
        }
        *(v4f*)&sQK[wrow * SROW + lane * 4] = ssum;      // reuse as GEMM2 A-matrix
        if (lane < NSLOT)
            __builtin_nontemporal_store(nao[lane], out + gb * OUTD + 2 * D + NSLOT * D + lane);
    }
    __syncthreads();

    // ---- reload weight buffer with WfT (RT: shared by all blocks) ----
    {
        const uint4* src = (const uint4*)wsWfT;
        uint4* dst = (uint4*)sW;
#pragma unroll
        for (int i = 0; i < 16; ++i) dst[tid + 128 * i] = src[tid + 128 * i];
    }
    __syncthreads();

    // ================= GEMM2: summary = s @ Wf^T (f16 WMMA) =================
    {
        v8f zero = {};
        v8f acc[8];
#pragma unroll
        for (int nt = 0; nt < 8; ++nt) acc[nt] = zero;
        const int arow = wr0 + (lane & 15);
#pragma unroll
        for (int kk = 0; kk < 4; ++kk) {
            const int kbase = kk * 32 + ((lane >= 16) ? 8 : 0);
            const float* ap = &sQK[arow * SROW + kbase];
            v16h a;
#pragma unroll
            for (int e = 0; e < 8; ++e) a[e] = (_Float16)ap[e];
#pragma unroll
            for (int e = 0; e < 8; ++e) a[8 + e] = (_Float16)ap[16 + e];
            const _Float16* bp = &sW[(kk * 32 + ((lane >= 16) ? 8 : 0)) * D + (lane & 15)];
#pragma unroll
            for (int nt = 0; nt < 8; ++nt) {
                v16h b;
#pragma unroll
                for (int e = 0; e < 8; ++e) b[e] = bp[e * D + nt * 16];
#pragma unroll
                for (int e = 0; e < 8; ++e) b[8 + e] = bp[(16 + e) * D + nt * 16];
                acc[nt] = __builtin_amdgcn_wmma_f32_16x16x32_f16(
                    false, a, false, b, (short)0, acc[nt], false, false);
            }
        }
        const int cn = lane & 15;
        const int mb = (lane >> 4) * 8;
#pragma unroll
        for (int nt = 0; nt < 8; ++nt)
#pragma unroll
            for (int j = 0; j < 8; ++j)
                __builtin_nontemporal_store(
                    acc[nt][j], out + (size_t)(gr0 + mb + j) * OUTD + D + nt * 16 + cn);
    }
}

// ---------------------------------------------------------------------------
extern "C" void kernel_launch(void* const* d_in, const int* in_sizes, int n_in,
                              void* d_out, int out_size, void* d_ws, size_t ws_size,
                              hipStream_t stream) {
    (void)n_in; (void)out_size; (void)ws_size;
    const float* state = (const float*)d_in[0];
    const float* slots = (const float*)d_in[1];
    const float* acts  = (const float*)d_in[2];
    const float* Wq    = (const float*)d_in[3];
    const float* Wk    = (const float*)d_in[4];
    const float* Wg_w  = (const float*)d_in[5];
    const float* Wg_b  = (const float*)d_in[6];
    const float* Wf    = (const float*)d_in[7];
    float* out = (float*)d_out;

    _Float16* wsM   = (_Float16*)d_ws;       // 128*128 f16
    _Float16* wsWfT = wsM + D * D;           // 128*128 f16  (64 KB total)

    const int B = in_sizes[0] / D;

    wm_prep<<<dim3((D * D) / 256), 256, 0, stream>>>(Wq, Wk, Wf, wsM, wsWfT);
    wm_main<<<dim3(B / ROWS_PER_BLOCK), 128, 0, stream>>>(
        state, slots, acts, Wg_w, Wg_b, wsM, wsWfT, out);
}
